// NCDE_63393717289486
// MI455X (gfx1250) — matrix operand order repaired and checked by reading, not compiled, not run on hardware
//
#include <hip/hip_runtime.h>

typedef __attribute__((ext_vector_type(16))) _Float16 v16h;
typedef __attribute__((ext_vector_type(8)))  float    v8f;

constexpr int NB  = 2048;   // batch
constexpr int NT  = 32;     // knots
constexpr int NC  = 16;     // input channels
constexpr int NH  = 64;     // hidden channels
constexpr int NS  = 128;    // hidden size
constexpr int NHC = NH * NC;   // 1024

// ---------------------------------------------------------------------------
// Pack a row-major [K,N] fp32 weight matrix into WMMA B-fragment layout, f16.
// Fragment for tile (ntile, kblk): element (lane, e) = W[kblk*32 + (lane>>4)*16 + e,
//                                                       ntile*16 + (lane&15)]
// Stored at dst[((ntile*KB + kblk)*32 + lane)*16 + e]  (KB = K/32)
// => each lane's 16 halves are contiguous (32B), one b128-pair load.
// ---------------------------------------------------------------------------
__global__ __launch_bounds__(256) void ncde_pack(const float* __restrict__ W,
                                                 _Float16* __restrict__ dst,
                                                 int K, int N) {
  int idx = blockIdx.x * 256 + threadIdx.x;
  int total = (K >> 5) * (N >> 4) * 512;
  if (idx >= total) return;
  int e     = idx & 15;
  int lane  = (idx >> 4) & 31;
  int tile  = idx >> 9;
  int KB    = K >> 5;
  int kblk  = tile % KB;
  int ntile = tile / KB;
  int k = kblk * 32 + (lane >> 4) * 16 + e;
  int n = ntile * 16 + (lane & 15);
  dst[idx] = (_Float16)W[k * N + n];
}

// Build a 16x32 f16 A-fragment from an LDS row (row-major activations).
// Layout (matches packed B): elem e -> K = kb*32 + (lane>>4)*16 + e.
template <typename T>
__device__ __forceinline__ v16h load_afrag(const T* row, int kb, int gh) {
  v16h a;
  #pragma unroll
  for (int e = 0; e < 8; ++e) a[e]     = (_Float16)row[kb * 32 + gh * 16 + e];
  #pragma unroll
  for (int e = 0; e < 8; ++e) a[8 + e] = (_Float16)row[kb * 32 + gh * 16 + 8 + e];
  return a;
}

// ---------------------------------------------------------------------------
// Main NCDE scan kernel. One block = 256 threads (8 wave32) = 16 batch rows.
// Grid = 128 workgroups -> fills the WGP array.
// ---------------------------------------------------------------------------
__global__ __launch_bounds__(256) void ncde_main(
    const float* __restrict__ x, const float* __restrict__ Winit,
    const float* __restrict__ binit,
    const _Float16* __restrict__ W1p, const float* __restrict__ b1,
    const _Float16* __restrict__ W2p, const float* __restrict__ b2,
    const _Float16* __restrict__ W3p, const float* __restrict__ b3,
    float* __restrict__ out)
{
  __shared__ float    zbuf[16][NH];   // current z (f32)
  __shared__ float    kacc[16][NH];   // k1 + 2k2 + 2k3 (+k4)
  __shared__ float    gbuf[16][NH];   // g(zs, s)
  __shared__ _Float16 zsh[16][NH];    // RK4 stage input, f16 (WMMA A precision)
  __shared__ _Float16 h1[16][NS];
  __shared__ _Float16 h2[16][NS];
  __shared__ float    dxs[3][16][NC]; // dX/dt at s = 0, 0.5, 1

  const int tid  = threadIdx.x;
  const int lane = tid & 31;
  const int wv   = tid >> 5;    // wave id 0..7
  const int gh   = lane >> 4;   // lane half
  const int ln   = lane & 15;
  const int bg   = blockIdx.x * 16;

  // ---- z0 = x[:,0,:] @ Winit + binit ----
  for (int idx = tid; idx < 16 * NH; idx += 256) {
    int b = idx >> 6, h = idx & 63;
    float s = binit[h];
    const float* xr = x + (size_t)(bg + b) * NT * NC;
    #pragma unroll
    for (int c = 0; c < NC; ++c) s += xr[c] * Winit[c * NH + h];
    zbuf[b][h] = s;
    zsh[b][h]  = (_Float16)s;
  }

  // ---- loop-invariant per-lane registers ----
  v16h w1f[2], w2f[4];
  #pragma unroll
  for (int kb = 0; kb < 2; ++kb)
    w1f[kb] = *(const v16h*)(W1p + ((size_t)(wv * 2 + kb) * 32 + lane) * 16);
  #pragma unroll
  for (int kb = 0; kb < 4; ++kb)
    w2f[kb] = *(const v16h*)(W2p + ((size_t)(wv * 4 + kb) * 32 + lane) * 16);
  const float b1v = b1[wv * 16 + ln];
  const float b2v = b2[wv * 16 + ln];
  float b3v[8];
  #pragma unroll
  for (int it = 0; it < 8; ++it)
    b3v[it] = b3[(wv * 8 + it) * 16 + ln];

  __syncthreads();

  for (int i = 0; i < NT - 1; ++i) {
    // ---- dX/dt at the three s values ----
    for (int idx = tid; idx < 16 * NC; idx += 256) {
      int b = idx >> 4, c = idx & 15;
      const float* xr = x + (size_t)(bg + b) * NT * NC + c;
      float xi   = xr[i * NC];
      float xip1 = xr[(i + 1) * NC];
      int   im   = (i == 0) ? 0 : (i - 1);
      float xm   = xr[im * NC];
      float dip1 = xip1 - xi;
      float di   = (i == 0) ? dip1 : (xi - xm);
      dxs[0][b][c] = di;                                                   // s = 0
      dxs[1][b][c] = -1.5f * xi - 0.25f * di + 1.5f * xip1 - 0.25f * dip1; // s = 0.5
      dxs[2][b][c] = dip1;                                                 // s = 1
    }
    __syncthreads();

    #pragma unroll 1
    for (int r = 0; r < 4; ++r) {
      const int sidx = (r == 0) ? 0 : ((r == 3) ? 2 : 1);

      // ======== stage 1: h1 = relu(zs @ W1 + b1); wave owns N-tile wv ========
      {
        v16h a[2];                      // issue all A loads, then WMMA chain
        #pragma unroll
        for (int kb = 0; kb < 2; ++kb) a[kb] = load_afrag(zsh[ln], kb, gh);
        v8f acc = {};
        #pragma unroll
        for (int kb = 0; kb < 2; ++kb)
          acc = __builtin_amdgcn_wmma_f32_16x16x32_f16(false, a[kb], false, w1f[kb],
                                                       (short)0, acc, false, false);
        #pragma unroll
        for (int j = 0; j < 8; ++j) {
          float v = acc[j] + b1v;
          h1[j + 8 * gh][wv * 16 + ln] = (_Float16)(v > 0.f ? v : 0.f);
        }
      }
      __syncthreads();

      // ======== stage 2: h2 = relu(h1 @ W2 + b2) ========
      {
        v16h a[4];
        #pragma unroll
        for (int kb = 0; kb < 4; ++kb) a[kb] = load_afrag(h1[ln], kb, gh);
        v8f acc = {};
        #pragma unroll
        for (int kb = 0; kb < 4; ++kb)
          acc = __builtin_amdgcn_wmma_f32_16x16x32_f16(false, a[kb], false, w2f[kb],
                                                       (short)0, acc, false, false);
        #pragma unroll
        for (int j = 0; j < 8; ++j) {
          float v = acc[j] + b2v;
          h2[j + 8 * gh][wv * 16 + ln] = (_Float16)(v > 0.f ? v : 0.f);
        }
      }
      __syncthreads();

      // ======== stage 3: y = h2 @ W3 + b3, fused einsum over C ========
      // Each 16x16 D-tile (ntile nt) is exactly y[., h=nt, c=0..15]:
      // g[b, nt] = sum_c (y + b3) * dX[b, c]  -> 16-lane butterfly reduce.
      {
        v16h a3[4];
        #pragma unroll
        for (int kb = 0; kb < 4; ++kb) a3[kb] = load_afrag(h2[ln], kb, gh);
        float dxv[8];
        #pragma unroll
        for (int j = 0; j < 8; ++j)
          dxv[j] = dxs[sidx][j + 8 * gh][ln];

        #pragma unroll 2                // interleave two N-tiles' loads + WMMAs
        for (int it = 0; it < 8; ++it) {
          int nt = wv * 8 + it;
          const v16h* bp = (const v16h*)W3p + ((size_t)nt * 4) * 32 + lane;
          if (it + 1 < 8)               // stream next W3 tile from L2 ahead of use
            __builtin_prefetch((const v16h*)W3p + ((size_t)(nt + 1) * 4) * 32 + lane, 0, 1);
          v16h bf[4];
          #pragma unroll
          for (int kb = 0; kb < 4; ++kb) bf[kb] = bp[kb * 32];
          v8f acc = {};
          #pragma unroll
          for (int kb = 0; kb < 4; ++kb)
            acc = __builtin_amdgcn_wmma_f32_16x16x32_f16(false, a3[kb], false, bf[kb],
                                                         (short)0, acc, false, false);
          #pragma unroll
          for (int j = 0; j < 8; ++j) {
            float v = (acc[j] + b3v[it]) * dxv[j];
            v += __shfl_xor(v, 1);
            v += __shfl_xor(v, 2);
            v += __shfl_xor(v, 4);
            v += __shfl_xor(v, 8);
            if (ln == 0) gbuf[j + 8 * gh][nt] = v;
          }
        }
      }
      __syncthreads();

      // ======== RK4 combine ========
      for (int idx = tid; idx < 16 * NH; idx += 256) {
        int b = idx >> 6, h = idx & 63;
        float kk = gbuf[b][h];
        if (r == 0)      { kacc[b][h] = kk;         zsh[b][h] = (_Float16)(zbuf[b][h] + 0.5f * kk); }
        else if (r == 1) { kacc[b][h] += 2.f * kk;  zsh[b][h] = (_Float16)(zbuf[b][h] + 0.5f * kk); }
        else if (r == 2) { kacc[b][h] += 2.f * kk;  zsh[b][h] = (_Float16)(zbuf[b][h] + kk); }
        else {
          float zn = zbuf[b][h] + (kacc[b][h] + kk) * (1.f / 6.f);
          zbuf[b][h] = zn;
          zsh[b][h]  = (_Float16)zn;   // zs = z for next interval's k1
        }
      }
      __syncthreads();
    }
  }

  for (int idx = tid; idx < 16 * NH; idx += 256) {
    int b = idx >> 6, h = idx & 63;
    out[(size_t)(bg + b) * NH + h] = zbuf[b][h];
  }
}

// ---------------------------------------------------------------------------
extern "C" void kernel_launch(void* const* d_in, const int* in_sizes, int n_in,
                              void* d_out, int out_size, void* d_ws, size_t ws_size,
                              hipStream_t stream) {
  const float* x     = (const float*)d_in[0];
  const float* Winit = (const float*)d_in[1];
  const float* binit = (const float*)d_in[2];
  const float* W1    = (const float*)d_in[3];
  const float* b1    = (const float*)d_in[4];
  const float* W2    = (const float*)d_in[5];
  const float* b2    = (const float*)d_in[6];
  const float* W3    = (const float*)d_in[7];
  const float* b3    = (const float*)d_in[8];

  _Float16* W1p = (_Float16*)d_ws;       // 64x128   ->  8192 halves
  _Float16* W2p = W1p + NH * NS;         // 128x128  -> 16384 halves
  _Float16* W3p = W2p + NS * NS;         // 128x1024 -> 131072 halves (304KB total)

  ncde_pack<<<(NH * NS  + 255) / 256, 256, 0, stream>>>(W1, W1p, NH, NS);
  ncde_pack<<<(NS * NS  + 255) / 256, 256, 0, stream>>>(W2, W2p, NS, NS);
  ncde_pack<<<(NS * NHC + 255) / 256, 256, 0, stream>>>(W3, W3p, NS, NHC);

  ncde_main<<<NB / 16, 256, 0, stream>>>(x, Winit, binit,
                                         W1p, b1, W2p, b2, W3p, b3,
                                         (float*)d_out);
}
